// GPTNeoSelfAttention_27453430956166
// MI455X (gfx1250) — compile-verified, matrix operand order checked by reading, not run on hardware
//
#include <hip/hip_runtime.h>
#include <hip/hip_bf16.h>
#include <stdint.h>

// ---------------------------------------------------------------------------
// GPT-Neo local self-attention, MI455X (gfx1250, wave32, WMMA).
//
// Pipeline (all on `stream`):
//   1) cvt fp32->bf16:  X, Wq, Wk, Wv, Wo
//   2) gemm_nt (bf16 WMMA, async global->LDS double-buffered staging):
//        Q = X@Wq^T  -> [B,H,S,D] bf16
//        K = X@Wk^T  -> [B,H,S,D] bf16
//        V = X@Wv^T  -> [B,H,D,S] bf16 (transposed for the P*V GEMM)
//   3) attn: flash-style banded softmax (WINDOW=256), WMMA for QK^T and P*V,
//            ctx -> [B,S,E] bf16 (reuses Xbf region)
//   4) gemm_nt: out = ctx@Wo^T + bo -> fp32 d_out
// ---------------------------------------------------------------------------

#define BATCH 4
#define SQ    2048
#define EMB   2048
#define NH    16
#define DH    128
#define WIN   256

typedef float  v8f   __attribute__((ext_vector_type(8)));
typedef __bf16 v16bf __attribute__((ext_vector_type(16)));
typedef __bf16 v8bf  __attribute__((ext_vector_type(8)));

// ---------------------------------------------------------------------------
// Async global->LDS 16B copy (GLOBAL_LOAD_ASYNC_TO_LDS_B128, ASYNCcnt).
// Builtin signature (from clang diagnostic): pointee is a GCC vector
// 'int __attribute__((vector_size(16)))', source in addrspace(1) (__device__),
// dest in addrspace(3). Falls back to register staging if unavailable.
// ---------------------------------------------------------------------------
#if __has_builtin(__builtin_amdgcn_global_load_async_to_lds_b128)
#define HAVE_ASYNC_LDS 1
#else
#define HAVE_ASYNC_LDS 0
#endif

typedef int v4i_t __attribute__((vector_size(16)));
typedef __attribute__((address_space(1))) v4i_t* gv4p;
typedef __attribute__((address_space(3))) v4i_t* lv4p;

__device__ __forceinline__ void copy16_g2l(const __bf16* g, __bf16* l) {
#if HAVE_ASYNC_LDS
  // generic LDS address carries the DS byte offset in bits [31:0]
  __builtin_amdgcn_global_load_async_to_lds_b128(
      (gv4p)(uintptr_t)g, (lv4p)(uintptr_t)(uint32_t)(uintptr_t)l, 0, 0);
#else
  *(v8bf*)l = *(const v8bf*)g;
#endif
}

__device__ __forceinline__ void wait_async_copies() {
#if HAVE_ASYNC_LDS
#if __has_builtin(__builtin_amdgcn_s_wait_asynccnt)
  __builtin_amdgcn_s_wait_asynccnt(0);
#else
  asm volatile("s_wait_asynccnt 0" ::: "memory");
#endif
#endif
}

// ---------------------------------------------------------------------------
// Fragment loader for the NT layout (A 16x32 MxK, or B 32x16 KxN where each
// "column" n is a contiguous row of the source matrix).
// Per ISA 16-bit A layout: lanes 0-15 hold row r, K = {0..7, 16..23};
// lanes 16-31 hold row r, K = {8..15, 24..31}.  Two 16B loads per lane.
// ---------------------------------------------------------------------------
__device__ __forceinline__ v16bf load_frag(const __bf16* __restrict__ base,
                                           int ld, int lane) {
  const int r  = lane & 15;
  const int hs = (lane >> 4) << 3;  // 0 or 8 halves
  union { v16bf v; v8bf h[2]; } u;
  u.h[0] = *(const v8bf*)(base + (size_t)r * ld + hs);
  u.h[1] = *(const v8bf*)(base + (size_t)r * ld + hs + 16);
  return u.v;
}

// ---------------------------------------------------------------------------
// fp32 -> bf16 conversion, 8 elements/thread (b128 in/out)
// ---------------------------------------------------------------------------
__global__ __launch_bounds__(256)
void cvt_kernel(const float* __restrict__ src, __bf16* __restrict__ dst, int n8) {
  const int i = blockIdx.x * 256 + threadIdx.x;
  if (i >= n8) return;
  const float4* s = (const float4*)src + (size_t)i * 2;
  const float4 a = s[0], b = s[1];
  v8bf o;
  o[0] = (__bf16)a.x; o[1] = (__bf16)a.y; o[2] = (__bf16)a.z; o[3] = (__bf16)a.w;
  o[4] = (__bf16)b.x; o[5] = (__bf16)b.y; o[6] = (__bf16)b.z; o[7] = (__bf16)b.w;
  *((v8bf*)dst + i) = o;
}

// ---------------------------------------------------------------------------
// NT GEMM: C[m,n] = sum_k A[m,k] * B[n,k]   (A: [M,K] bf16, B: [N,K] bf16)
// Block tile 128x128, 8 waves, wave tile 32x64 (2x4 WMMA), K-step 32,
// double-buffered async global->LDS staging.
// mode 0: bf16 out, head-split [B,H,S,D]
// mode 2: bf16 out, head-split transposed [B,H,D,S]
// mode 3: fp32 out [M,N] row-major + bias[n]
// ---------------------------------------------------------------------------
#define LDT 40  // LDS tile row stride in halves (32 + 8 pad)

__global__ __launch_bounds__(256)
void gemm_nt(const __bf16* __restrict__ A, const __bf16* __restrict__ Bm,
             __bf16* __restrict__ dstb, float* __restrict__ dstf,
             const float* __restrict__ bias, int mode, int Kdim) {
  __shared__ __bf16 sA[2][128 * LDT];
  __shared__ __bf16 sB[2][128 * LDT];
  const int n0   = blockIdx.x * 128;
  const int m0   = blockIdx.y * 128;
  const int tid  = threadIdx.x;
  const int lane = tid & 31;
  const int wid  = tid >> 5;
  const int wm   = wid >> 1;   // 0..3 -> 32 rows each
  const int wn   = wid & 1;    // 0..1 -> 64 cols each

  v8f acc[2][4] = {};

  const int lr = tid >> 2;         // 0..63
  const int lc = (tid & 3) << 3;   // 0,8,16,24

  auto stage = [&](int buf, int k0) {
    copy16_g2l(&A[(size_t)(m0 + lr) * Kdim + k0 + lc],       &sA[buf][lr * LDT + lc]);
    copy16_g2l(&A[(size_t)(m0 + lr + 64) * Kdim + k0 + lc],  &sA[buf][(lr + 64) * LDT + lc]);
    copy16_g2l(&Bm[(size_t)(n0 + lr) * Kdim + k0 + lc],      &sB[buf][lr * LDT + lc]);
    copy16_g2l(&Bm[(size_t)(n0 + lr + 64) * Kdim + k0 + lc], &sB[buf][(lr + 64) * LDT + lc]);
  };

  // prologue: fill buffer 0
  stage(0, 0);
  wait_async_copies();
  __syncthreads();

  for (int k0 = 0; k0 < Kdim; k0 += 32) {
    const int cur = (k0 >> 5) & 1;
    if (k0 + 32 < Kdim) stage(cur ^ 1, k0 + 32);  // prefetch next tile

    v16bf af[2], bf[4];
#pragma unroll
    for (int i = 0; i < 2; ++i)
      af[i] = load_frag(&sA[cur][(wm * 32 + i * 16) * LDT], LDT, lane);
#pragma unroll
    for (int j = 0; j < 4; ++j)
      bf[j] = load_frag(&sB[cur][(wn * 64 + j * 16) * LDT], LDT, lane);
#pragma unroll
    for (int i = 0; i < 2; ++i)
#pragma unroll
      for (int j = 0; j < 4; ++j)
        acc[i][j] = __builtin_amdgcn_wmma_f32_16x16x32_bf16(
            false, af[i], false, bf[j], (short)0, acc[i][j], false, false);

    // our ds_loads from buf[cur] must retire before anyone overwrites it,
    // and our async copies into buf[cur^1] must land before anyone reads it
    asm volatile("s_wait_dscnt 0" ::: "memory");
    wait_async_copies();
    __syncthreads();
  }

  // Epilogue. C-frag layout: element (m = v + 8*(lane>>4), n = lane&15).
  const int mbase = (lane >> 4) * 8;
  const int ncol  = lane & 15;
#pragma unroll
  for (int i = 0; i < 2; ++i)
#pragma unroll
    for (int j = 0; j < 4; ++j)
#pragma unroll
      for (int v = 0; v < 8; ++v) {
        const int row = m0 + wm * 32 + i * 16 + mbase + v;
        const int col = n0 + wn * 64 + j * 16 + ncol;
        const float val = acc[i][j][v];
        if (mode == 3) {
          dstf[(size_t)row * EMB + col] = val + bias[col];
        } else {
          const int b = row >> 11, s = row & (SQ - 1);
          const int h = col >> 7,  d = col & (DH - 1);
          size_t idx;
          if (mode == 2) idx = ((size_t)(b * NH + h) * DH + d) * SQ + s;   // V^T
          else           idx = ((size_t)(b * NH + h) * SQ + s) * DH + d;   // Q/K
          dstb[idx] = (__bf16)val;
        }
      }
}

// ---------------------------------------------------------------------------
// Banded flash attention. Block = 1 (b,h,64-row query tile), 4 waves.
// Each wave owns 16 query rows independently (no cross-wave sync needed).
// Key tiles of 64, processed diagonal-first so the running max is valid
// before any fully-masked row can appear.
// ---------------------------------------------------------------------------
#define LDP 72  // P-tile row stride in halves (64 + 8 pad)

__global__ __launch_bounds__(128)
void attn_kernel(const __bf16* __restrict__ Qb, const __bf16* __restrict__ Kb,
                 const __bf16* __restrict__ Vt, __bf16* __restrict__ ctx) {
  __shared__ __bf16 sP[4][16 * LDP];

  const int tid  = threadIdx.x;
  const int lane = tid & 31;
  const int wid  = tid >> 5;
  const int qt   = blockIdx.x & (SQ / 64 - 1);   // 32 query tiles
  const int bh   = blockIdx.x >> 5;
  const int b    = bh >> 4, h = bh & (NH - 1);
  const int r0   = qt * 64 + wid * 16;           // this wave's 16 query rows

  const __bf16* Q  = Qb + (size_t)bh * SQ * DH;  // [S,D]
  const __bf16* Kp = Kb + (size_t)bh * SQ * DH;  // [S,D]
  const __bf16* Vp = Vt + (size_t)bh * DH * SQ;  // [D,S]

  // Q fragments for this wave's rows (D = 128 -> 4 K-steps)
  v16bf qf[4];
#pragma unroll
  for (int d = 0; d < 4; ++d)
    qf[d] = load_frag(Q + (size_t)r0 * DH + d * 32, DH, lane);

  v8f o[8] = {};
  float mrow[8], lrow[8];
#pragma unroll
  for (int v = 0; v < 8; ++v) { mrow[v] = -1e30f; lrow[v] = 0.f; }

  const int half8 = (lane >> 4) * 8;
  const int ln    = lane & 15;
  const int tLast = r0 >> 6;
  int jmin = r0 - (WIN - 1); if (jmin < 0) jmin = 0;
  const int tFirst = jmin >> 6;

  for (int t = tLast; t >= tFirst; --t) {
    const int j0 = t << 6;

    // ---- S = Q K^T (16 x 64, fp32) ----
    v8f sacc[4] = {};
#pragma unroll
    for (int nt = 0; nt < 4; ++nt)
#pragma unroll
      for (int d = 0; d < 4; ++d) {
        v16bf kf = load_frag(Kp + (size_t)(j0 + nt * 16) * DH + d * 32, DH, lane);
        sacc[nt] = __builtin_amdgcn_wmma_f32_16x16x32_bf16(
            false, qf[d], false, kf, (short)0, sacc[nt], false, false);
      }

    // ---- band mask + per-tile row max ----
    float tmax[8];
#pragma unroll
    for (int v = 0; v < 8; ++v) tmax[v] = -1e30f;
#pragma unroll
    for (int nt = 0; nt < 4; ++nt)
#pragma unroll
      for (int v = 0; v < 8; ++v) {
        const int i = r0 + half8 + v;
        const int j = j0 + nt * 16 + ln;
        float s = sacc[nt][v];
        const bool ok = (j <= i) && ((i - j) < WIN);
        s = ok ? s : -1e30f;
        sacc[nt][v] = s;
        tmax[v] = fmaxf(tmax[v], s);
      }

    // ---- online softmax update (row = 16 lanes of one half) ----
#pragma unroll
    for (int v = 0; v < 8; ++v) {
#pragma unroll
      for (int m = 1; m < 16; m <<= 1)
        tmax[v] = fmaxf(tmax[v], __shfl_xor(tmax[v], m, 32));
      const float mn    = fmaxf(mrow[v], tmax[v]);
      const float alpha = __expf(mrow[v] - mn);
      mrow[v] = mn;
      lrow[v] *= alpha;
#pragma unroll
      for (int ct = 0; ct < 8; ++ct) o[ct][v] *= alpha;
    }

    // ---- P = exp(S - m): row sums + stage to LDS in row-major ----
    float tsum[8];
#pragma unroll
    for (int v = 0; v < 8; ++v) tsum[v] = 0.f;
#pragma unroll
    for (int nt = 0; nt < 4; ++nt)
#pragma unroll
      for (int v = 0; v < 8; ++v) {
        const float p = __expf(sacc[nt][v] - mrow[v]);
        tsum[v] += p;
        sP[wid][(half8 + v) * LDP + nt * 16 + ln] = (__bf16)p;
      }
#pragma unroll
    for (int v = 0; v < 8; ++v) {
#pragma unroll
      for (int m = 1; m < 16; m <<= 1) tsum[v] += __shfl_xor(tsum[v], m, 32);
      lrow[v] += tsum[v];
    }

    // wave-private LDS region; DS ops are in-order per wave, barrier the
    // compiler + outstanding DS stores before re-reading as A-fragments
    asm volatile("s_wait_dscnt 0" ::: "memory");

    // ---- O += P * V  (A = P 16x64, B rows = V^T rows, contiguous in j) ----
#pragma unroll
    for (int kk = 0; kk < 2; ++kk) {
      v16bf pf = load_frag(&sP[wid][kk * 32], LDP, lane);
#pragma unroll
      for (int ct = 0; ct < 8; ++ct) {
        v16bf vf = load_frag(Vp + (size_t)(ct * 16) * SQ + j0 + kk * 32, SQ, lane);
        o[ct] = __builtin_amdgcn_wmma_f32_16x16x32_bf16(
            false, pf, false, vf, (short)0, o[ct], false, false);
      }
    }
    asm volatile("s_wait_dscnt 0" ::: "memory");
  }

  // ---- normalize + write ctx [B,S,E] bf16 ----
#pragma unroll
  for (int v = 0; v < 8; ++v) {
    const float inv = 1.0f / lrow[v];
    const int i = r0 + half8 + v;
#pragma unroll
    for (int ct = 0; ct < 8; ++ct) {
      const int d = ct * 16 + ln;
      ctx[(size_t)(b * SQ + i) * EMB + h * DH + d] = (__bf16)(o[ct][v] * inv);
    }
  }
}

// ---------------------------------------------------------------------------
extern "C" void kernel_launch(void* const* d_in, const int* in_sizes, int n_in,
                              void* d_out, int out_size, void* d_ws, size_t ws_size,
                              hipStream_t stream) {
  const float* X  = (const float*)d_in[0];
  const float* Wq = (const float*)d_in[1];
  const float* Wk = (const float*)d_in[2];
  const float* Wv = (const float*)d_in[3];
  const float* Wo = (const float*)d_in[4];
  const float* bo = (const float*)d_in[5];
  float* out = (float*)d_out;
  char*  ws  = (char*)d_ws;

  const size_t SZ_X = (size_t)BATCH * SQ * EMB * sizeof(__bf16);  // 32 MiB
  const size_t SZ_W = (size_t)EMB * EMB * sizeof(__bf16);         //  8 MiB

  __bf16* Xbf = (__bf16*)(ws);
  __bf16* Wqb = (__bf16*)(ws + SZ_X);
  __bf16* Wkb = (__bf16*)(ws + SZ_X + 1 * SZ_W);
  __bf16* Wvb = (__bf16*)(ws + SZ_X + 2 * SZ_W);
  __bf16* Wob = (__bf16*)(ws + SZ_X + 3 * SZ_W);
  __bf16* Qbf = (__bf16*)(ws + SZ_X + 4 * SZ_W);
  __bf16* Kbf = (__bf16*)(ws + 2 * SZ_X + 4 * SZ_W);
  __bf16* Vtb = (__bf16*)(ws + 3 * SZ_X + 4 * SZ_W);   // total 160 MiB
  __bf16* Ctx = Xbf;  // X is dead after the QKV GEMMs; reuse its region

  const int nX = BATCH * SQ * EMB;  // 16777216
  const int nW = EMB * EMB;         //  4194304

  cvt_kernel<<<nX / 8 / 256, 256, 0, stream>>>(X,  Xbf, nX / 8);
  cvt_kernel<<<nW / 8 / 256, 256, 0, stream>>>(Wq, Wqb, nW / 8);
  cvt_kernel<<<nW / 8 / 256, 256, 0, stream>>>(Wk, Wkb, nW / 8);
  cvt_kernel<<<nW / 8 / 256, 256, 0, stream>>>(Wv, Wvb, nW / 8);
  cvt_kernel<<<nW / 8 / 256, 256, 0, stream>>>(Wo, Wob, nW / 8);

  dim3 gg(EMB / 128, BATCH * SQ / 128);  // (16, 64)
  gemm_nt<<<gg, 256, 0, stream>>>(Xbf, Wqb, Qbf, nullptr, nullptr, 0, EMB);
  gemm_nt<<<gg, 256, 0, stream>>>(Xbf, Wkb, Kbf, nullptr, nullptr, 0, EMB);
  gemm_nt<<<gg, 256, 0, stream>>>(Xbf, Wvb, Vtb, nullptr, nullptr, 2, EMB);

  attn_kernel<<<BATCH * NH * (SQ / 64), 128, 0, stream>>>(Qbf, Kbf, Vtb, Ctx);

  gemm_nt<<<gg, 256, 0, stream>>>(Ctx, Wob, nullptr, out, bo, 3, EMB);
}
